// Seq2SeqLSTM_72679436582937
// MI455X (gfx1250) — compile-verified, matrix operand order checked by reading, not run on hardware
//
#include <hip/hip_runtime.h>
#include <hip/hip_bf16.h>

#define BB   64
#define SS   256
#define TT   64
#define EE   128
#define HH   256
#define AA   256
#define VV   32000
#define NPB  250           // proj blocks (32000/128)

typedef __attribute__((ext_vector_type(16))) __bf16 v16bf;
typedef __attribute__((ext_vector_type(8)))  __bf16 v8bf;
typedef __attribute__((ext_vector_type(8)))  float  v8f;

__device__ __forceinline__ float sigf(float x) { return 1.f / (1.f + __expf(-x)); }

// Load a 16x32 bf16 fragment (rows x K) from row-major [row][ldk] storage.
// lane L: row = L&15 ; vector elems 0..7 = k 8h+0..7, elems 8..15 = k 16+8h+0..7 (h = L>>4)
__device__ __forceinline__ v16bf load_frag(const __bf16* p0, int ldk, int lane) {
  int r  = lane & 15;
  int h8 = (lane >> 4) & 1;
  const __bf16* p = p0 + (size_t)r * ldk + 8 * h8;
  v8bf lo = *(const v8bf*)(p);
  v8bf hi = *(const v8bf*)(p + 16);
  v16bf out;
#pragma unroll
  for (int i = 0; i < 8; ++i) { out[i] = lo[i]; out[8 + i] = hi[i]; }
  return out;
}

__device__ __forceinline__ v8f wmma_bf16(v16bf a, v16bf b, v8f c) {
  return __builtin_amdgcn_wmma_f32_16x16x32_bf16(false, a, false, b, (short)0, c, false, false);
}

// One LSTM layer step for a 16-row batch slice. 8 waves; wave w owns h-chunks
// {16w, 16w+128}. A1: [16 x K1] bf16 (row-major, stride ld1). hbuf: LDS bf16
// [16*HH], read as recurrent input then overwritten with new h. creg/hout:
// per-thread cell state / new h, layout [ch*8 + r].
template <int K1>
__device__ __forceinline__ void lstm_layer(const __bf16* A1, int ld1, __bf16* hbuf,
                                           const __bf16* Wih, const __bf16* Whh,
                                           const float* bias, float* creg, float* hout,
                                           int wave, int lane) {
  v8f acc[2][4];
#pragma unroll
  for (int ch = 0; ch < 2; ++ch)
#pragma unroll
    for (int g = 0; g < 4; ++g)
#pragma unroll
      for (int r = 0; r < 8; ++r) acc[ch][g][r] = 0.f;

  for (int kf = 0; kf < K1 / 32; ++kf) {
    v16bf a = load_frag(A1 + kf * 32, ld1, lane);
#pragma unroll
    for (int ch = 0; ch < 2; ++ch) {
      int hb = 16 * (wave + 8 * ch);
#pragma unroll
      for (int g = 0; g < 4; ++g) {
        int n0 = g * HH + hb;
        v16bf w = load_frag(Wih + (size_t)n0 * K1 + kf * 32, K1, lane);
        acc[ch][g] = wmma_bf16(a, w, acc[ch][g]);
      }
    }
  }
  for (int kf = 0; kf < HH / 32; ++kf) {
    v16bf a = load_frag(hbuf + kf * 32, HH, lane);
#pragma unroll
    for (int ch = 0; ch < 2; ++ch) {
      int hb = 16 * (wave + 8 * ch);
#pragma unroll
      for (int g = 0; g < 4; ++g) {
        int n0 = g * HH + hb;
        v16bf w = load_frag(Whh + (size_t)n0 * HH + kf * 32, HH, lane);
        acc[ch][g] = wmma_bf16(a, w, acc[ch][g]);
      }
    }
  }
  __syncthreads();  // all waves done reading hbuf
  int half = lane >> 4;
#pragma unroll
  for (int ch = 0; ch < 2; ++ch) {
    int hb = 16 * (wave + 8 * ch);
    int n  = hb + (lane & 15);
    float bi = bias[n], bf_ = bias[HH + n], bg = bias[2 * HH + n], bo = bias[3 * HH + n];
#pragma unroll
    for (int r = 0; r < 8; ++r) {
      int m   = r + 8 * half;
      float ig = sigf(acc[ch][0][r] + bi);
      float fg = sigf(acc[ch][1][r] + bf_);
      float gg = tanhf(acc[ch][2][r] + bg);
      float og = sigf(acc[ch][3][r] + bo);
      float c  = fg * creg[ch * 8 + r] + ig * gg;
      creg[ch * 8 + r] = c;
      float h = og * tanhf(c);
      hout[ch * 8 + r] = h;
      hbuf[m * HH + n] = (__bf16)h;
    }
  }
  __syncthreads();  // new h visible to next consumer
}

// ---------------- prep kernels ----------------
__global__ void cvt_kernel(const float* src, __bf16* dst, int n) {
  int i = blockIdx.x * 256 + threadIdx.x;
  if (i < n) dst[i] = (__bf16)src[i];
}

__global__ void embed_kernel(const int* x, const float* Ein, __bf16* emb) {
  size_t i = (size_t)blockIdx.x * 256 + threadIdx.x;  // over S*B*E
  int e = (int)(i & (EE - 1));
  size_t sb = i >> 7;
  int b = (int)(sb & (BB - 1));
  int s = (int)(sb >> 6);
  int tk = x[(size_t)b * SS + s];
  emb[i] = (__bf16)Ein[(size_t)tk * EE + e];
}

// ---------------- encoder: 4 blocks x 256 threads, sequential over S ----------------
__global__ __launch_bounds__(256) void encoder_kernel(const __bf16* emb, const int* x,
                                                      const __bf16* Wih0, const __bf16* Whh0, const float* b0,
                                                      const __bf16* Wih1, const __bf16* Whh1, const float* b1,
                                                      float* henc, __bf16* hencB) {
  __shared__ __align__(16) __bf16 h0s[16 * HH];
  __shared__ __align__(16) __bf16 h1s[16 * HH];
  int bb = blockIdx.x, tid = threadIdx.x, wave = tid >> 5, lane = tid & 31;
  for (int i = tid; i < 16 * HH; i += 256) { h0s[i] = (__bf16)0.f; h1s[i] = (__bf16)0.f; }
  float c0[16], c1[16], hout[16];
#pragma unroll
  for (int i = 0; i < 16; ++i) { c0[i] = 0.f; c1[i] = 0.f; }
  __syncthreads();

  for (int s = 0; s < SS; ++s) {
    const __bf16* A = emb + ((size_t)(s * BB + bb * 16)) * EE;
    lstm_layer<EE>(A, EE, h0s, Wih0, Whh0, b0, c0, hout, wave, lane);
    lstm_layer<HH>(h0s, HH, h1s, Wih1, Whh1, b1, c1, hout, wave, lane);
    int half = lane >> 4;
#pragma unroll
    for (int ch = 0; ch < 2; ++ch) {
      int n = 16 * (wave + 8 * ch) + (lane & 15);
#pragma unroll
      for (int r = 0; r < 8; ++r) {
        int m = r + 8 * half;
        int b = bb * 16 + m;
        float h = (x[(size_t)b * SS + s] != 0) ? hout[ch * 8 + r] : 0.f;
        size_t off = ((size_t)b * SS + s) * HH + n;
        henc[off]  = h;
        hencB[off] = (__bf16)h;
      }
    }
    __syncthreads();
  }
}

// ---------------- k_enc = henc @ Wk.T + bk : [B*S, A] WMMA GEMM ----------------
__global__ __launch_bounds__(256) void kenc_kernel(const __bf16* hencB, const __bf16* WkB,
                                                   const float* bk, float* kenc) {
  int row0 = blockIdx.x * 64;
  int wave = threadIdx.x >> 5, lane = threadIdx.x & 31;
  int half = lane >> 4;
  for (int mt = 0; mt < 4; ++mt) {
#pragma unroll
    for (int w2 = 0; w2 < 2; ++w2) {
      int nt = wave * 2 + w2;
      v8f acc;
#pragma unroll
      for (int r = 0; r < 8; ++r) acc[r] = 0.f;
      for (int kf = 0; kf < HH / 32; ++kf) {
        v16bf a = load_frag(hencB + (size_t)(row0 + mt * 16) * HH + kf * 32, HH, lane);
        v16bf b = load_frag(WkB + (size_t)(nt * 16) * HH + kf * 32, HH, lane);
        acc = wmma_bf16(a, b, acc);
      }
      int n = nt * 16 + (lane & 15);
      float bv = bk[n];
#pragma unroll
      for (int r = 0; r < 8; ++r) {
        int m = row0 + mt * 16 + r + 8 * half;
        kenc[(size_t)m * AA + n] = acc[r] + bv;
      }
    }
  }
}

// ---------------- decoder per-step kernels ----------------
__global__ void dec_init_kernel(float* c0, float* c1, __bf16* h0, __bf16* h1, int* tok) {
  int i = blockIdx.x * 256 + threadIdx.x;
  if (i < BB * HH) { c0[i] = 0.f; c1[i] = 0.f; h0[i] = (__bf16)0.f; h1[i] = (__bf16)0.f; }
  if (i < BB) tok[i] = 1;  // BOS
}

__global__ __launch_bounds__(256) void dec_attn_kernel(const __bf16* h1g, const float* Wq,
                                                       const float* bq, const float* kenc,
                                                       const float* henc, const float* vvec,
                                                       const int* x, const int* tok,
                                                       const float* Eout, __bf16* xin) {
  __shared__ float qs[AA];
  __shared__ float ws_[SS];
  __shared__ float red[SS];
  int b = blockIdx.x, tid = threadIdx.x;
  // q = h1 @ Wq.T + bq
  float acc = bq[tid];
  for (int k = 0; k < HH; ++k) acc += (float)h1g[(size_t)b * HH + k] * Wq[(size_t)tid * HH + k];
  qs[tid] = acc;
  __syncthreads();
  // score_s = v . tanh(q + k_enc[b,s,:]) , masked
  const float* kb = kenc + ((size_t)b * SS + tid) * AA;
  float sc = 0.f;
  for (int a = 0; a < AA; ++a) sc += vvec[a] * tanhf(qs[a] + kb[a]);
  if (x[(size_t)b * SS + tid] == 0) sc = -1e9f;
  red[tid] = sc;
  __syncthreads();
  for (int off = 128; off > 0; off >>= 1) { if (tid < off) red[tid] = fmaxf(red[tid], red[tid + off]); __syncthreads(); }
  float mx = red[0];
  __syncthreads();
  float e = __expf(sc - mx);
  ws_[tid] = e;
  red[tid] = e;
  __syncthreads();
  for (int off = 128; off > 0; off >>= 1) { if (tid < off) red[tid] += red[tid + off]; __syncthreads(); }
  float inv = 1.f / red[0];
  __syncthreads();
  // ctx[h] = sum_s w_s * henc[b,s,h]
  float c = 0.f;
  for (int s2 = 0; s2 < SS; ++s2) c += ws_[s2] * henc[((size_t)b * SS + s2) * HH + tid];
  c *= inv;
  int tk = tok[b];
  if (tid < EE) xin[(size_t)b * (EE + HH) + tid] = (__bf16)Eout[(size_t)tk * EE + tid];
  xin[(size_t)b * (EE + HH) + EE + tid] = (__bf16)c;
}

__global__ __launch_bounds__(256) void dec_lstm_kernel(const __bf16* xin,
                                                       const __bf16* Wih0, const __bf16* Whh0, const float* b0,
                                                       const __bf16* Wih1, const __bf16* Whh1, const float* b1,
                                                       float* c0g, float* c1g, __bf16* h0g, __bf16* h1g) {
  __shared__ __align__(16) __bf16 h0s[16 * HH];
  __shared__ __align__(16) __bf16 h1s[16 * HH];
  int bb = blockIdx.x, tid = threadIdx.x, wave = tid >> 5, lane = tid & 31;
  for (int i = tid; i < 16 * HH; i += 256) { h0s[i] = h0g[(size_t)bb * 16 * HH + i]; h1s[i] = h1g[(size_t)bb * 16 * HH + i]; }
  float c0[16], c1[16], hout0[16], hout1[16];
  int half = lane >> 4;
#pragma unroll
  for (int ch = 0; ch < 2; ++ch) {
    int n = 16 * (wave + 8 * ch) + (lane & 15);
#pragma unroll
    for (int r = 0; r < 8; ++r) {
      int b = bb * 16 + r + 8 * half;
      c0[ch * 8 + r] = c0g[(size_t)b * HH + n];
      c1[ch * 8 + r] = c1g[(size_t)b * HH + n];
    }
  }
  __syncthreads();
  lstm_layer<EE + HH>(xin + (size_t)bb * 16 * (EE + HH), EE + HH, h0s, Wih0, Whh0, b0, c0, hout0, wave, lane);
  lstm_layer<HH>(h0s, HH, h1s, Wih1, Whh1, b1, c1, hout1, wave, lane);
#pragma unroll
  for (int ch = 0; ch < 2; ++ch) {
    int n = 16 * (wave + 8 * ch) + (lane & 15);
#pragma unroll
    for (int r = 0; r < 8; ++r) {
      int b = bb * 16 + r + 8 * half;
      c0g[(size_t)b * HH + n] = c0[ch * 8 + r];
      c1g[(size_t)b * HH + n] = c1[ch * 8 + r];
    }
  }
  for (int i = tid; i < 16 * HH; i += 256) { h0g[(size_t)bb * 16 * HH + i] = h0s[i]; h1g[(size_t)bb * 16 * HH + i] = h1s[i]; }
}

// logits = h1 @ Wproj.T + bproj -> d_out[:, t, :], plus per-block argmax partials
__global__ __launch_bounds__(256) void dec_proj_kernel(const __bf16* h1B, const __bf16* WprojB,
                                                       const float* bproj, float* out_t,
                                                       float* pval, int* pidx) {
  __shared__ float lg[64][128];
  int blk = blockIdx.x;
  int n_base = blk * 128;
  int wave = threadIdx.x >> 5, lane = threadIdx.x & 31;
  int half = lane >> 4;
  int nt = wave;  // 8 n-tiles per block, one per wave
  for (int mt = 0; mt < 4; ++mt) {
    v8f acc;
#pragma unroll
    for (int r = 0; r < 8; ++r) acc[r] = 0.f;
    for (int kf = 0; kf < HH / 32; ++kf) {
      v16bf a = load_frag(h1B + (size_t)(mt * 16) * HH + kf * 32, HH, lane);
      v16bf b = load_frag(WprojB + (size_t)(n_base + nt * 16) * HH + kf * 32, HH, lane);
      acc = wmma_bf16(a, b, acc);
    }
    int nl = nt * 16 + (lane & 15);
    int n  = n_base + nl;
    float bv = bproj[n];
#pragma unroll
    for (int r = 0; r < 8; ++r) {
      int m = mt * 16 + r + 8 * half;
      float vv = acc[r] + bv;
      lg[m][nl] = vv;
      out_t[(size_t)m * ((size_t)TT * VV) + n] = vv;
    }
  }
  __syncthreads();
  if (threadIdx.x < 64) {
    int b = threadIdx.x;
    float best = lg[b][0];
    int bi = n_base;
    for (int j = 1; j < 128; ++j) { float v2 = lg[b][j]; if (v2 > best) { best = v2; bi = n_base + j; } }
    pval[b * NPB + blk] = best;
    pidx[b * NPB + blk] = bi;
  }
}

__global__ void dec_argmax_kernel(const float* pval, const int* pidx, int* tok) {
  int b = threadIdx.x;
  if (b >= BB) return;
  float best = pval[b * NPB];
  int bi = pidx[b * NPB];
  for (int j = 1; j < NPB; ++j) {
    float v2 = pval[b * NPB + j];
    if (v2 > best) { best = v2; bi = pidx[b * NPB + j]; }
  }
  tok[b] = bi;
}

extern "C" void kernel_launch(void* const* d_in, const int* in_sizes, int n_in,
                              void* d_out, int out_size, void* d_ws, size_t ws_size,
                              hipStream_t stream) {
  const int*   x     = (const int*)d_in[0];
  const float* Ein   = (const float*)d_in[2];
  const float* Eout  = (const float*)d_in[3];
  const float* eWih0 = (const float*)d_in[4];
  const float* eWhh0 = (const float*)d_in[5];
  const float* eb0   = (const float*)d_in[6];
  const float* eWih1 = (const float*)d_in[7];
  const float* eWhh1 = (const float*)d_in[8];
  const float* eb1   = (const float*)d_in[9];
  const float* dWih0 = (const float*)d_in[10];
  const float* dWhh0 = (const float*)d_in[11];
  const float* db0   = (const float*)d_in[12];
  const float* dWih1 = (const float*)d_in[13];
  const float* dWhh1 = (const float*)d_in[14];
  const float* db1   = (const float*)d_in[15];
  const float* Wq    = (const float*)d_in[16];
  const float* bq    = (const float*)d_in[17];
  const float* Wk    = (const float*)d_in[18];
  const float* bk    = (const float*)d_in[19];
  const float* vvec  = (const float*)d_in[20];
  const float* Wproj = (const float*)d_in[21];
  const float* bproj = (const float*)d_in[22];
  float* out = (float*)d_out;

  char* p = (char*)d_ws;
  auto alloc = [&p](size_t bytes) { char* r = p; p += (bytes + 255) & ~(size_t)255; return r; };

  __bf16* eWih0B = (__bf16*)alloc((size_t)1024 * 128 * 2);
  __bf16* eWhh0B = (__bf16*)alloc((size_t)1024 * 256 * 2);
  __bf16* eWih1B = (__bf16*)alloc((size_t)1024 * 256 * 2);
  __bf16* eWhh1B = (__bf16*)alloc((size_t)1024 * 256 * 2);
  __bf16* dWih0B = (__bf16*)alloc((size_t)1024 * 384 * 2);
  __bf16* dWhh0B = (__bf16*)alloc((size_t)1024 * 256 * 2);
  __bf16* dWih1B = (__bf16*)alloc((size_t)1024 * 256 * 2);
  __bf16* dWhh1B = (__bf16*)alloc((size_t)1024 * 256 * 2);
  __bf16* WkB    = (__bf16*)alloc((size_t)256 * 256 * 2);
  __bf16* WprojB = (__bf16*)alloc((size_t)VV * 256 * 2);
  __bf16* embB   = (__bf16*)alloc((size_t)SS * BB * EE * 2);
  float*  henc   = (float*)alloc((size_t)BB * SS * HH * 4);
  __bf16* hencB  = (__bf16*)alloc((size_t)BB * SS * HH * 2);
  float*  kenc   = (float*)alloc((size_t)BB * SS * AA * 4);
  float*  c0g    = (float*)alloc((size_t)BB * HH * 4);
  float*  c1g    = (float*)alloc((size_t)BB * HH * 4);
  __bf16* h0g    = (__bf16*)alloc((size_t)BB * HH * 2);
  __bf16* h1g    = (__bf16*)alloc((size_t)BB * HH * 2);
  __bf16* xinB   = (__bf16*)alloc((size_t)BB * (EE + HH) * 2);
  float*  pval   = (float*)alloc((size_t)BB * NPB * 4);
  int*    pidx   = (int*)alloc((size_t)BB * NPB * 4);
  int*    tok    = (int*)alloc((size_t)BB * 4);

  auto cvt = [&](const float* s, __bf16* d, int n) {
    cvt_kernel<<<(n + 255) / 256, 256, 0, stream>>>(s, d, n);
  };
  cvt(eWih0, eWih0B, 1024 * 128);
  cvt(eWhh0, eWhh0B, 1024 * 256);
  cvt(eWih1, eWih1B, 1024 * 256);
  cvt(eWhh1, eWhh1B, 1024 * 256);
  cvt(dWih0, dWih0B, 1024 * 384);
  cvt(dWhh0, dWhh0B, 1024 * 256);
  cvt(dWih1, dWih1B, 1024 * 256);
  cvt(dWhh1, dWhh1B, 1024 * 256);
  cvt(Wk, WkB, 256 * 256);
  cvt(Wproj, WprojB, VV * 256);

  embed_kernel<<<(SS * BB * EE) / 256, 256, 0, stream>>>(x, Ein, embB);

  encoder_kernel<<<4, 256, 0, stream>>>(embB, x, eWih0B, eWhh0B, eb0,
                                        eWih1B, eWhh1B, eb1, henc, hencB);

  kenc_kernel<<<(BB * SS) / 64, 256, 0, stream>>>(hencB, WkB, bk, kenc);

  dec_init_kernel<<<(BB * HH + 255) / 256, 256, 0, stream>>>(c0g, c1g, h0g, h1g, tok);

  for (int t = 0; t < TT; ++t) {
    dec_attn_kernel<<<BB, 256, 0, stream>>>(h1g, Wq, bq, kenc, henc, vvec, x, tok, Eout, xinB);
    dec_lstm_kernel<<<4, 256, 0, stream>>>(xinB, dWih0B, dWhh0B, db0,
                                           dWih1B, dWhh1B, db1, c0g, c1g, h0g, h1g);
    dec_proj_kernel<<<NPB, 256, 0, stream>>>(h1g, WprojB, bproj, out + (size_t)t * VV, pval, pidx);
    dec_argmax_kernel<<<1, 64, 0, stream>>>(pval, pidx, tok);
  }
}
